// Decoder_25709674234511
// MI455X (gfx1250) — compile-verified, matrix operand order checked by reading, not run on hardware
//
#include <hip/hip_runtime.h>
#include <math.h>

// ---------------- Problem constants (fixed by the reference) ----------------
#define Bc    4
#define Sc    1024
#define SEc   1024
#define DM    1024
#define Hc    16
#define DKH   64
#define Lc    6
#define DFFC  4096
#define MTOT  (Bc * Sc)          // 4096 rows in every activation GEMM

// ---------------- WMMA types / helpers ----------------
typedef __attribute__((ext_vector_type(16))) _Float16 v16h;
typedef __attribute__((ext_vector_type(8)))  _Float16 v8h;
typedef __attribute__((ext_vector_type(8)))  float    v8f;

__device__ __forceinline__ unsigned short f2h(float f) {
    _Float16 h = (_Float16)f;
    return __builtin_bit_cast(unsigned short, h);
}
__device__ __forceinline__ ushort4 f2h4(float4 f) {
    ushort4 u;
    u.x = f2h(f.x); u.y = f2h(f.y); u.z = f2h(f.z); u.w = f2h(f.w);
    return u;
}

// Load a 16x32 (A) or [N=16]x[K=32] (B) f16 fragment from LDS.
// Lane mapping per CDNA5 ISA 7.12.2 (16-bit matrices):
// lanes 0-15 hold K {0..7, 16..23}, lanes 16-31 hold K {8..15, 24..31}.
__device__ __forceinline__ v16h load_frag(const unsigned short* p0, int stride, int lane) {
    const int half = (lane >> 4) & 1;
    const unsigned short* p = p0 + (lane & 15) * stride + half * 8;
    v8h lo = *reinterpret_cast<const v8h*>(p);
    v8h hi = *reinterpret_cast<const v8h*>(p + 16);
    v16h r;
#pragma unroll
    for (int i = 0; i < 8; ++i) { r[i] = lo[i]; r[i + 8] = hi[i]; }
    return r;
}

__device__ __forceinline__ v8f wmma16(v16h a, v16h b, v8f c) {
    return __builtin_amdgcn_wmma_f32_16x16x32_f16(false, a, false, b, (short)0, c,
                                                  false, false);
}

// ---------------- Block mean/rstd reduction (256 threads = 8 waves) ----------
__device__ __forceinline__ void block_stats(float s, float s2, float* red,
                                            float& mu, float& inv) {
    const int lane = threadIdx.x & 31;
    const int wv = threadIdx.x >> 5;
#pragma unroll
    for (int o = 16; o; o >>= 1) {
        s  += __shfl_xor(s,  o, 32);
        s2 += __shfl_xor(s2, o, 32);
    }
    if (lane == 0) { red[wv] = s; red[8 + wv] = s2; }
    __syncthreads();
    float ts = 0.f, ts2 = 0.f;
#pragma unroll
    for (int w = 0; w < 8; ++w) { ts += red[w]; ts2 += red[8 + w]; }
    mu = ts / (float)DM;
    float var = ts2 / (float)DM - mu * mu;
    inv = rsqrtf(var + 1e-5f);
    __syncthreads();
}

// ---------------- Embedding + sinusoidal PE + LayerNorm(ln0) ----------------
__global__ __launch_bounds__(256) void embed_ln0_kernel(
    const int* __restrict__ dec, const float* __restrict__ emb,
    const float* __restrict__ g, const float* __restrict__ bb,
    float* __restrict__ y) {
    __shared__ float xb[DM];
    __shared__ float red[16];
    const int t = blockIdx.x;
    const int pos = t & (Sc - 1);
    const int tok = dec[t];
    float s = 0.f, s2 = 0.f;
    for (int i = threadIdx.x; i < DM; i += 256) {
        const int k2 = i & ~1;
        const float div = expf(-9.210340371976184f * (float)k2 / (float)DM);
        const float a = (float)pos * div;
        const float pe = (i & 1) ? cosf(a) : sinf(a);
        const float v = emb[(size_t)tok * DM + i] + pe;
        xb[i] = v; s += v; s2 += v * v;
    }
    __syncthreads();
    float mu, inv;
    block_stats(s, s2, red, mu, inv);
    for (int i = threadIdx.x; i < DM; i += 256)
        y[(size_t)t * DM + i] = (xb[i] - mu) * inv * g[i] + bb[i];
}

// ---------------- Generic LayerNorm over D=1024 ----------------
__global__ __launch_bounds__(256) void layernorm_kernel(
    const float* __restrict__ x, const float* __restrict__ g,
    const float* __restrict__ bb, float* __restrict__ y) {
    __shared__ float red[16];
    const size_t row = blockIdx.x;
    const float* xr = x + row * DM;
    float s = 0.f, s2 = 0.f;
    for (int i = threadIdx.x; i < DM; i += 256) {
        const float v = xr[i]; s += v; s2 += v * v;
    }
    float mu, inv;
    block_stats(s, s2, red, mu, inv);
    for (int i = threadIdx.x; i < DM; i += 256)
        y[row * DM + i] = (xr[i] - mu) * inv * g[i] + bb[i];
}

// ---------------- WMMA GEMM: C = f(A*B + bias) [+ resid] ----------------
// A: [M,K] f32 row-major. B: bT ? [N,K] : [K,N] f32 row-major.
// Block tile 128x128, 8 waves (4x2), each wave 32x64 (acc[2][4]), BK=64.
// Software-pipelined: tile k+1 global loads issue under tile k WMMA compute.
#define BM  128
#define BN  128
#define BKT 64
#define AST 72
#define BST 72
__global__ __launch_bounds__(256) void gemm_kernel(
    const float* __restrict__ A, const float* __restrict__ Bm,
    const float* __restrict__ bias, const float* __restrict__ resid,
    float* __restrict__ C, int M, int N, int K, int bT, int relu) {
    __shared__ __align__(16) unsigned short As[BM * AST];  // 18 KB
    __shared__ __align__(16) unsigned short Bs[BN * BST];  // 18 KB
    const int tid = threadIdx.x;
    const int lane = tid & 31;
    const int wave = tid >> 5;
    const int wm = (wave >> 1) * 32;   // 0,32,64,96
    const int wn = (wave & 1) * 64;    // 0,64
    const int m0 = blockIdx.y * BM;
    const int n0 = blockIdx.x * BN;
    v8f acc[2][4] = {};

    ushort4 aH[8], bH[8];
    // ---- prologue fetch (tile k0 = 0), converted to f16 at fetch time ----
#pragma unroll
    for (int i = 0; i < 8; ++i) {
        const int idx = tid + i * 256;
        const int r = idx >> 4, c4 = (idx & 15) << 2;
        aH[i] = f2h4(*reinterpret_cast<const float4*>(A + (size_t)(m0 + r) * K + c4));
    }
    if (bT) {
#pragma unroll
        for (int i = 0; i < 8; ++i) {
            const int idx = tid + i * 256;
            const int r = idx >> 4, c4 = (idx & 15) << 2;
            bH[i] = f2h4(*reinterpret_cast<const float4*>(Bm + (size_t)(n0 + r) * K + c4));
        }
    } else {
#pragma unroll
        for (int i = 0; i < 8; ++i) {
            const int idx = tid + i * 256;
            const int r = idx >> 5, c4 = (idx & 31) << 2;
            bH[i] = f2h4(*reinterpret_cast<const float4*>(Bm + (size_t)r * N + n0 + c4));
        }
    }

    for (int k0 = 0; k0 < K; k0 += BKT) {
        __syncthreads();
        // ---- store current tile registers -> LDS ----
#pragma unroll
        for (int i = 0; i < 8; ++i) {
            const int idx = tid + i * 256;
            const int r = idx >> 4, c4 = (idx & 15) << 2;
            *reinterpret_cast<ushort4*>(As + r * AST + c4) = aH[i];
        }
        if (bT) {
#pragma unroll
            for (int i = 0; i < 8; ++i) {
                const int idx = tid + i * 256;
                const int r = idx >> 4, c4 = (idx & 15) << 2;
                *reinterpret_cast<ushort4*>(Bs + r * BST + c4) = bH[i];
            }
        } else {
#pragma unroll
            for (int i = 0; i < 8; ++i) {
                const int idx = tid + i * 256;
                const int r = idx >> 5, c4 = (idx & 31) << 2;
                Bs[(c4 + 0) * BST + r] = bH[i].x;
                Bs[(c4 + 1) * BST + r] = bH[i].y;
                Bs[(c4 + 2) * BST + r] = bH[i].z;
                Bs[(c4 + 3) * BST + r] = bH[i].w;
            }
        }
        __syncthreads();
        // ---- issue next tile's global loads (overlap with WMMA below) ----
        const int kn = k0 + BKT;
        if (kn < K) {
#pragma unroll
            for (int i = 0; i < 8; ++i) {
                const int idx = tid + i * 256;
                const int r = idx >> 4, c4 = (idx & 15) << 2;
                aH[i] = f2h4(*reinterpret_cast<const float4*>(
                    A + (size_t)(m0 + r) * K + kn + c4));
            }
            if (bT) {
#pragma unroll
                for (int i = 0; i < 8; ++i) {
                    const int idx = tid + i * 256;
                    const int r = idx >> 4, c4 = (idx & 15) << 2;
                    bH[i] = f2h4(*reinterpret_cast<const float4*>(
                        Bm + (size_t)(n0 + r) * K + kn + c4));
                }
            } else {
#pragma unroll
                for (int i = 0; i < 8; ++i) {
                    const int idx = tid + i * 256;
                    const int r = idx >> 5, c4 = (idx & 31) << 2;
                    bH[i] = f2h4(*reinterpret_cast<const float4*>(
                        Bm + (size_t)(kn + r) * N + n0 + c4));
                }
            }
            if (kn + BKT < K)   // L2 warm-up for the tile after next
                __builtin_prefetch(A + (size_t)(m0 + (tid >> 1)) * K + kn + BKT, 0, 1);
        }
        // ---- WMMA on current LDS tiles: 16 WMMAs per k-step ----
#pragma unroll
        for (int kk = 0; kk < BKT; kk += 32) {
            const v16h a0 = load_frag(As + (wm +  0) * AST + kk, AST, lane);
            const v16h a1 = load_frag(As + (wm + 16) * AST + kk, AST, lane);
#pragma unroll
            for (int j = 0; j < 4; ++j) {
                const v16h b = load_frag(Bs + (wn + j * 16) * BST + kk, BST, lane);
                acc[0][j] = wmma16(a0, b, acc[0][j]);
                acc[1][j] = wmma16(a1, b, acc[1][j]);
            }
        }
    }
    // ---- epilogue ----
    const int half = lane >> 4;
#pragma unroll
    for (int i = 0; i < 2; ++i) {
#pragma unroll
        for (int j = 0; j < 4; ++j) {
            const int col = n0 + wn + j * 16 + (lane & 15);
            const float bv = bias ? bias[col] : 0.f;
#pragma unroll
            for (int r8 = 0; r8 < 8; ++r8) {
                const int row = m0 + wm + i * 16 + r8 + 8 * half;
                float v = acc[i][j][r8] + bv;
                if (relu) v = fmaxf(v, 0.f);
                if (resid) v += resid[(size_t)row * N + col];
                C[(size_t)row * N + col] = v;
            }
        }
    }
}

// ---------------- Flash attention (WMMA, online softmax) ----------------
// grid: (B*H, Sq/128), 256 threads = 8 waves, each wave = 16 query rows.
// K/V tile kt+32 is fetched while tile kt is being computed.
__global__ __launch_bounds__(256) void attn_kernel(
    const float* __restrict__ Q, const float* __restrict__ Kb,
    const float* __restrict__ Vb, const int* __restrict__ mids,
    float* __restrict__ Ctx, float* __restrict__ sout,
    int Sq, int Sk, int causal) {
    const int bh = blockIdx.x;
    const int bIdx = bh / Hc;
    const int hIdx = bh % Hc;
    const int tid = threadIdx.x;
    const int lane = tid & 31;
    const int wave = tid >> 5;
    const int half = lane >> 4;
    const int q0 = blockIdx.y * 128;
    const int qw = q0 + wave * 16;

    __shared__ __align__(16) unsigned short Qs[128 * 72]; // 128 q rows x 64 feat
    __shared__ __align__(16) unsigned short Ks[32 * 72];  // 32 keys  x 64 feat
    __shared__ __align__(16) unsigned short Vt[64 * 40];  // 64 feat  x 32 keys
    __shared__ __align__(16) unsigned short Ps[8 * 16 * 40]; // per-wave P 16x32

    // ---- Q tile: fetch all -> convert -> store (two-phase) ----
    ushort4 qH[8];
#pragma unroll
    for (int i = 0; i < 8; ++i) {
        const int idx = tid + i * 256;
        const int r = idx >> 4, c4 = (idx & 15) << 2;
        qH[i] = f2h4(*reinterpret_cast<const float4*>(
            Q + ((size_t)bIdx * Sq + q0 + r) * DM + hIdx * DKH + c4));
    }
#pragma unroll
    for (int i = 0; i < 8; ++i) {
        const int idx = tid + i * 256;
        const int r = idx >> 4, c4 = (idx & 15) << 2;
        *reinterpret_cast<ushort4*>(Qs + r * 72 + c4) = qH[i];
    }
    __syncthreads();
    const v16h qa0 = load_frag(Qs + wave * 16 * 72 +  0, 72, lane);
    const v16h qa1 = load_frag(Qs + wave * 16 * 72 + 32, 72, lane);

    float m[8], ssum[8];
    v8f o[4] = {};
#pragma unroll
    for (int j = 0; j < 8; ++j) { m[j] = -1e30f; ssum[j] = 0.f; }

    // ---- prologue K/V fetch (kt = 0) ----
    ushort4 kH[2], vH[2];
#pragma unroll
    for (int i = 0; i < 2; ++i) {
        const int idx = tid + i * 256;
        const int r = idx >> 4, c4 = (idx & 15) << 2;
        const size_t src = ((size_t)bIdx * Sk + r) * DM + hIdx * DKH + c4;
        kH[i] = f2h4(*reinterpret_cast<const float4*>(Kb + src));
        vH[i] = f2h4(*reinterpret_cast<const float4*>(Vb + src));
    }

    for (int kt = 0; kt < Sk; kt += 32) {
        __syncthreads();
        // ---- store current K/V registers -> LDS (V transposed) ----
#pragma unroll
        for (int i = 0; i < 2; ++i) {
            const int idx = tid + i * 256;
            const int r = idx >> 4, c4 = (idx & 15) << 2;
            *reinterpret_cast<ushort4*>(Ks + r * 72 + c4) = kH[i];
            Vt[(c4 + 0) * 40 + r] = vH[i].x;
            Vt[(c4 + 1) * 40 + r] = vH[i].y;
            Vt[(c4 + 2) * 40 + r] = vH[i].z;
            Vt[(c4 + 3) * 40 + r] = vH[i].w;
        }
        __syncthreads();
        // ---- next-tile fetch overlaps the compute below ----
        if (kt + 32 < Sk) {
#pragma unroll
            for (int i = 0; i < 2; ++i) {
                const int idx = tid + i * 256;
                const int r = idx >> 4, c4 = (idx & 15) << 2;
                const size_t src =
                    ((size_t)bIdx * Sk + kt + 32 + r) * DM + hIdx * DKH + c4;
                kH[i] = f2h4(*reinterpret_cast<const float4*>(Kb + src));
                vH[i] = f2h4(*reinterpret_cast<const float4*>(Vb + src));
            }
        }
        // ---- scores: Q(16x64) x K^T(64x32) ----
        v8f sc0 = {}, sc1 = {};
        sc0 = wmma16(qa0, load_frag(Ks +  0 * 72 +  0, 72, lane), sc0);
        sc0 = wmma16(qa1, load_frag(Ks +  0 * 72 + 32, 72, lane), sc0);
        sc1 = wmma16(qa0, load_frag(Ks + 16 * 72 +  0, 72, lane), sc1);
        sc1 = wmma16(qa1, load_frag(Ks + 16 * 72 + 32, 72, lane), sc1);

        const int col0 = kt + (lane & 15);
        const int col1 = col0 + 16;
        const bool pad0 = (mids[(size_t)bIdx * Sk + col0] == 0);
        const bool pad1 = (mids[(size_t)bIdx * Sk + col1] == 0);
        unsigned short* pw = Ps + wave * 640;
#pragma unroll
        for (int j = 0; j < 8; ++j) {
            const int rg = qw + j + 8 * half;
            float s0 = sc0[j] * 0.125f;   // 1/sqrt(64)
            float s1 = sc1[j] * 0.125f;
            if (pad0 || (causal && col0 > rg)) s0 = -1e9f;
            if (pad1 || (causal && col1 > rg)) s1 = -1e9f;
            if (sout) {
                sout[((size_t)bh * Sq + rg) * Sk + col0] = s0;
                sout[((size_t)bh * Sq + rg) * Sk + col1] = s1;
            }
            float mx = fmaxf(s0, s1);
            mx = fmaxf(mx, __shfl_xor(mx, 1, 32));
            mx = fmaxf(mx, __shfl_xor(mx, 2, 32));
            mx = fmaxf(mx, __shfl_xor(mx, 4, 32));
            mx = fmaxf(mx, __shfl_xor(mx, 8, 32));
            const float mnew = fmaxf(m[j], mx);
            const float corr = expf(m[j] - mnew);
            m[j] = mnew;
            const float p0 = expf(s0 - mnew);
            const float p1 = expf(s1 - mnew);
            float rs = p0 + p1;
            rs += __shfl_xor(rs, 1, 32);
            rs += __shfl_xor(rs, 2, 32);
            rs += __shfl_xor(rs, 4, 32);
            rs += __shfl_xor(rs, 8, 32);
            ssum[j] = ssum[j] * corr + rs;
            o[0][j] *= corr; o[1][j] *= corr; o[2][j] *= corr; o[3][j] *= corr;
            pw[(j + 8 * half) * 40 + (lane & 15)]      = f2h(p0);
            pw[(j + 8 * half) * 40 + (lane & 15) + 16] = f2h(p1);
        }
        __syncthreads();
        // ---- O += P(16x32) x V(32x64) ----
        const v16h pa = load_frag(pw, 40, lane);
#pragma unroll
        for (int nt = 0; nt < 4; ++nt)
            o[nt] = wmma16(pa, load_frag(Vt + nt * 16 * 40, 40, lane), o[nt]);
    }
    // ---- normalize + write ctx ----
#pragma unroll
    for (int nt = 0; nt < 4; ++nt) {
        const int col = hIdx * DKH + nt * 16 + (lane & 15);
#pragma unroll
        for (int j = 0; j < 8; ++j) {
            const int rg = qw + j + 8 * half;
            Ctx[((size_t)bIdx * Sq + rg) * DM + col] = o[nt][j] / ssum[j];
        }
    }
}

// ---------------- Orchestration ----------------
extern "C" void kernel_launch(void* const* d_in, const int* in_sizes, int n_in,
                              void* d_out, int out_size, void* d_ws, size_t ws_size,
                              hipStream_t stream) {
    (void)in_sizes; (void)n_in; (void)out_size; (void)ws_size;
    const int*   enc_ids = (const int*)d_in[0];
    const int*   dec_ids = (const int*)d_in[1];
    const float* enc_out = (const float*)d_in[2];
    const float* emb     = (const float*)d_in[3];
    const float* ln0_g   = (const float*)d_in[4];
    const float* ln0_b   = (const float*)d_in[5];
    const float* sWq = (const float*)d_in[6],  *sWk = (const float*)d_in[7];
    const float* sWv = (const float*)d_in[8],  *sWo = (const float*)d_in[9];
    const float* sbq = (const float*)d_in[10], *sbk = (const float*)d_in[11];
    const float* sbv = (const float*)d_in[12], *sbo = (const float*)d_in[13];
    const float* cWq = (const float*)d_in[14], *cWk = (const float*)d_in[15];
    const float* cWv = (const float*)d_in[16], *cWo = (const float*)d_in[17];
    const float* cbq = (const float*)d_in[18], *cbk = (const float*)d_in[19];
    const float* cbv = (const float*)d_in[20], *cbo = (const float*)d_in[21];
    const float* fW1 = (const float*)d_in[22], *fb1 = (const float*)d_in[23];
    const float* fW2 = (const float*)d_in[24], *fb2 = (const float*)d_in[25];
    const float* l1g = (const float*)d_in[26], *l1b = (const float*)d_in[27];
    const float* l2g = (const float*)d_in[28], *l2b = (const float*)d_in[29];
    const float* l3g = (const float*)d_in[30], *l3b = (const float*)d_in[31];

    float* ws = (float*)d_ws;
    const size_t TD = (size_t)MTOT * DM;          // 4M floats per buffer
    float* hB = ws;            // residual stream
    float* bA = ws + 1 * TD;   // LN outputs / residual source
    float* bB = ws + 2 * TD;   // post-attention sums
    float* bQ = ws + 3 * TD;
    float* bK = ws + 4 * TD;
    float* bV = ws + 5 * TD;
    float* bC = ws + 6 * TD;   // attention context
    float* bF = ws + 7 * TD;   // FF hidden: 4096 x 4096
    float* scores = (float*)d_out + TD;           // last_w lives in d_out tail

    embed_ln0_kernel<<<MTOT, 256, 0, stream>>>(dec_ids, emb, ln0_g, ln0_b, hB);

    const dim3 gN(DM / BN, MTOT / BM);     // (8, 32)
    const dim3 gF(DFFC / BN, MTOT / BM);   // (32, 32)
    const dim3 gA(Bc * Hc, Sc / 128);

    for (int l = 0; l < Lc; ++l) {
        const size_t wO = (size_t)l * DM * DM;
        const size_t bO = (size_t)l * DM;
        const size_t w1O = (size_t)l * DFFC * DM;
        const size_t b1O = (size_t)l * DFFC;
        const size_t w2O = (size_t)l * DM * DFFC;

        // ---- self attention block ----
        layernorm_kernel<<<MTOT, 256, 0, stream>>>(hB, l1g + bO, l1b + bO, bA);
        gemm_kernel<<<gN, 256, 0, stream>>>(bA, sWq + wO, sbq + bO, nullptr, bQ,
                                            MTOT, DM, DM, 0, 0);
        gemm_kernel<<<gN, 256, 0, stream>>>(bA, sWk + wO, sbk + bO, nullptr, bK,
                                            MTOT, DM, DM, 0, 0);
        gemm_kernel<<<gN, 256, 0, stream>>>(bA, sWv + wO, sbv + bO, nullptr, bV,
                                            MTOT, DM, DM, 0, 0);
        attn_kernel<<<gA, 256, 0, stream>>>(bQ, bK, bV, dec_ids, bC, nullptr,
                                            Sc, Sc, 1);
        gemm_kernel<<<gN, 256, 0, stream>>>(bC, sWo + wO, sbo + bO, bA, bB,
                                            MTOT, DM, DM, 0, 0);
        // ---- cross attention block ----
        layernorm_kernel<<<MTOT, 256, 0, stream>>>(bB, l2g + bO, l2b + bO, bA);
        gemm_kernel<<<gN, 256, 0, stream>>>(enc_out, cWk + wO, cbk + bO, nullptr,
                                            bK, MTOT, DM, DM, 0, 0);
        gemm_kernel<<<gN, 256, 0, stream>>>(enc_out, cWv + wO, cbv + bO, nullptr,
                                            bV, MTOT, DM, DM, 0, 0);
        gemm_kernel<<<gN, 256, 0, stream>>>(bA, cWq + wO, cbq + bO, nullptr, bQ,
                                            MTOT, DM, DM, 0, 0);
        attn_kernel<<<gA, 256, 0, stream>>>(bQ, bK, bV, enc_ids, bC,
                                            (l == Lc - 1) ? scores : nullptr,
                                            Sc, SEc, 0);
        gemm_kernel<<<gN, 256, 0, stream>>>(bC, cWo + wO, cbo + bO, bA, bB,
                                            MTOT, DM, DM, 0, 0);
        // ---- feed forward block ----
        layernorm_kernel<<<MTOT, 256, 0, stream>>>(bB, l3g + bO, l3b + bO, bA);
        gemm_kernel<<<gF, 256, 0, stream>>>(bA, fW1 + w1O, fb1 + b1O, nullptr, bF,
                                            MTOT, DFFC, DM, 1, 1);
        gemm_kernel<<<gN, 256, 0, stream>>>(bF, fW2 + w2O, fb2 + bO, bA, hB,
                                            MTOT, DM, DFFC, 1, 0);
    }
    hipMemcpyAsync(d_out, hB, TD * sizeof(float), hipMemcpyDeviceToDevice, stream);
}
